// MultiHeadAttention_35820027248842
// MI455X (gfx1250) — matrix operand order repaired and checked by reading, not compiled, not run on hardware
//
#include <hip/hip_runtime.h>

// ---------------------------------------------------------------------------
// MultiHeadAttention for MI455X (gfx1250, wave32, WMMA).
// All GEMMs use v_wmma_f32_16x16x32_bf16 (bf16 operands, f32 accum).
// GEMMs are register-blocked: one wave computes a 64x64 C macro-tile
// (4x4 WMMA tiles): 16 WMMAs per 8 fragment loads per K-step of 32.
// Wave-uniform indices are pinned to SGPRs via readfirstlane so epilogues
// use scalar branches and vectorized stores instead of exec juggling.
// ---------------------------------------------------------------------------

#define BATCH   2
#define DIMM    1024
#define SEQ     2048
#define NHEADS  16
#define DHEADD  64
#define INNERD  1024
#define QKVROWS 3072

typedef __attribute__((ext_vector_type(8)))  __bf16 v8bf;
typedef __attribute__((ext_vector_type(16))) __bf16 v16bf;
typedef __attribute__((ext_vector_type(8)))  float  v8f;

static __device__ __forceinline__ unsigned short f32_to_bf16(float f) {
  unsigned int u = __float_as_uint(f);
  u += 0x7FFFu + ((u >> 16) & 1u);     // round-to-nearest-even
  return (unsigned short)(u >> 16);
}

static __device__ __forceinline__ v8f wmma_bf16(v16bf a, v16bf b, v8f c) {
  return __builtin_amdgcn_wmma_f32_16x16x32_bf16(false, a, false, b,
                                                 (short)0, c, false, false);
}

static __device__ __forceinline__ v16bf join8(v8bf lo, v8bf hi) {
  v16bf r;
#pragma unroll
  for (int t = 0; t < 8; ++t) { r[t] = lo[t]; r[t + 8] = hi[t]; }
  return r;
}

// 64x64 C macro-tile per wave. A row-major [M][K]; B stored transposed
// (row-major [N][K]). acc[mi][ni] is the 16x16 tile at (m0+16mi, n0+16ni).
static __device__ __forceinline__ void mma_block64(const unsigned short* __restrict__ A,
                                                   size_t lda,
                                                   const unsigned short* __restrict__ Bt,
                                                   size_t ldb,
                                                   int K, int m0, int n0,
                                                   int lr, int half,
                                                   v8f acc[4][4]) {
  const unsigned short* ar[4];
  const unsigned short* br[4];
#pragma unroll
  for (int t = 0; t < 4; ++t) {
    ar[t] = A  + (size_t)(m0 + t * 16 + lr) * lda;
    br[t] = Bt + (size_t)(n0 + t * 16 + lr) * ldb;
  }
  for (int k = 0; k < K; k += 32) {
    v16bf a[4], b[4];
#pragma unroll
    for (int t = 0; t < 4; ++t) {
      a[t] = *reinterpret_cast<const v16bf*>(ar[t] + k + half * 16);
      b[t] = *reinterpret_cast<const v16bf*>(br[t] + k + half * 16);
    }
#pragma unroll
    for (int mi = 0; mi < 4; ++mi)
#pragma unroll
      for (int ni = 0; ni < 4; ++ni)
        acc[mi][ni] = wmma_bf16(a[mi], b[ni], acc[mi][ni]);
  }
}

// ---------------------------------------------------------------------------
// 1) elementwise fp32 -> bf16
__global__ void __launch_bounds__(256) k_cvt(const float* __restrict__ src,
                                             unsigned short* __restrict__ dst, int n) {
  int i = blockIdx.x * blockDim.x + threadIdx.x;
  if (i < n) dst[i] = f32_to_bf16(src[i]);
}

// 2) x[b][d][n] (fp32) -> xT[b][n][d] (bf16)
__global__ void __launch_bounds__(256) k_xT(const float* __restrict__ x,
                                            unsigned short* __restrict__ xT) {
  int i = blockIdx.x * blockDim.x + threadIdx.x;   // B*SEQ*DIM, d fastest
  int d = i & (DIMM - 1);
  int n = (i >> 10) & (SEQ - 1);
  int b = i >> 21;
  xT[i] = f32_to_bf16(x[((size_t)b * DIMM + d) * SEQ + n]);
}

// ---------------------------------------------------------------------------
// 3) QKV projection: qkv[o][n] = sum_d w_qkv[o][d] * x[d][n], per batch.
//    q,k -> [b][h][n][d] fp32 (contiguous v8f stores); v -> bf16 [b][h][d][n].
__global__ void __launch_bounds__(256) k_gemm_qkv(const unsigned short* __restrict__ wq,
                                                  const unsigned short* __restrict__ xT,
                                                  float* __restrict__ qT,
                                                  float* __restrict__ kT,
                                                  unsigned short* __restrict__ vb) {
  const int MT = QKVROWS / 64, NT = SEQ / 64;            // 48 x 32 macro-tiles
  int wid = __builtin_amdgcn_readfirstlane(
      blockIdx.x * ((int)blockDim.x >> 5) + ((int)threadIdx.x >> 5));
  int lane = threadIdx.x & 31, half = lane >> 4, lr = lane & 15;
  int b   = wid / (MT * NT);
  int rem = wid % (MT * NT);
  int m0  = (rem / NT) * 64, n0 = (rem % NT) * 64;

  v8f acc[4][4];
#pragma unroll
  for (int mi = 0; mi < 4; ++mi)
#pragma unroll
    for (int ni = 0; ni < 4; ++ni)
      acc[mi][ni] = (v8f){0.f,0.f,0.f,0.f,0.f,0.f,0.f,0.f};

  mma_block64(wq, DIMM, xT + (size_t)b * SEQ * DIMM, DIMM,
              DIMM, m0, n0, lr, half, acc);

  // 64-row M block lies entirely in one q/k/v section and one head (scalar).
  int sec = m0 >> 10;                    // 0=q, 1=k, 2=v
  int h   = (m0 & 1023) >> 6;
  size_t bh = (size_t)b * NHEADS + h;

  if (sec == 2) {
    // v: lane values hit different d rows of [d][n]; scalar bf16 stores.
#pragma unroll
    for (int mi = 0; mi < 4; ++mi)
#pragma unroll
      for (int ni = 0; ni < 4; ++ni) {
        int n = n0 + ni * 16 + lr;
#pragma unroll
        for (int r = 0; r < 8; ++r) {
          int d = mi * 16 + r + half * 8;
          vb[(bh * DHEADD + d) * SEQ + n] = f32_to_bf16(acc[mi][ni][r]);
        }
      }
  } else {
    // q/k: a lane's 8 slots are 8 consecutive d values -> one 32B store.
    float* dst = (sec == 0) ? qT : kT;
#pragma unroll
    for (int mi = 0; mi < 4; ++mi)
#pragma unroll
      for (int ni = 0; ni < 4; ++ni) {
        int n  = n0 + ni * 16 + lr;
        int d0 = mi * 16 + half * 8;
        *reinterpret_cast<v8f*>(dst + (bh * SEQ + n) * DHEADD + d0) = acc[mi][ni];
      }
  }
}

// ---------------------------------------------------------------------------
// 4) RoPE on q,k (layout [b][h][n][64]); q also scaled by DHEAD^-0.5; emit bf16.
__global__ void __launch_bounds__(256) k_rope(const float* __restrict__ qT,
                                              const float* __restrict__ kT,
                                              unsigned short* __restrict__ qb,
                                              unsigned short* __restrict__ kb) {
  int i = blockIdx.x * blockDim.x + threadIdx.x;   // B*H*SEQ*DHEAD = 2^22
  int d = i & 63;
  int n = (i >> 6) & (SEQ - 1);
  int r = d & 31;
  // inv_freq[r] = 10000^(-r/32); t = n * (8192/2048) = 4n
  float angle = __expf(-(float)r * (9.2103403719761836f / 32.0f)) * (4.0f * (float)n);
  float sn, cs;
  __sincosf(angle, &sn, &cs);
  int   partner = (d < 32) ? i + 32 : i - 32;
  float sign    = (d < 32) ? -1.0f : 1.0f;
  float qv = qT[i], qp = qT[partner];
  float kv = kT[i], kp = kT[partner];
  qb[i] = f32_to_bf16((qv * cs + sign * qp * sn) * 0.125f);
  kb[i] = f32_to_bf16( kv * cs + sign * kp * sn);
}

// ---------------------------------------------------------------------------
// 5) Flash attention. Block = 4 waves; wave owns 16 query rows of one (b,h).
//    Writes O^T as bf16 outT[b][n][h*64+d] (B-operand layout for out proj).
__global__ void __launch_bounds__(128) k_attn(const unsigned short* __restrict__ qb,
                                              const unsigned short* __restrict__ kb,
                                              const unsigned short* __restrict__ vb,
                                              unsigned short* __restrict__ outT) {
  __shared__ unsigned short lds[4][16 * 32];       // per-wave P tile (16 x 32)
  int wave = __builtin_amdgcn_readfirstlane((int)threadIdx.x >> 5);
  int lane = threadIdx.x & 31, half = lane >> 4, lr = lane & 15;
  int iblk = blockIdx.x & 31;                      // SEQ/64 = 32 query blocks
  int bh   = blockIdx.x >> 5;                      // B*H = 32
  int b = bh >> 4, h = bh & 15;
  int i0 = iblk * 64 + wave * 16;

  // Q fragments (K = d: two 16x32 A tiles), loaded once.
  const unsigned short* qrow = qb + ((size_t)bh * SEQ + i0 + lr) * DHEADD;
  v16bf aq0 = *(const v16bf*)(qrow +      half * 16);
  v16bf aq1 = *(const v16bf*)(qrow + 32 + half * 16);

  v8f acc0 = {0.f,0.f,0.f,0.f,0.f,0.f,0.f,0.f};
  v8f acc1 = acc0, acc2 = acc0, acc3 = acc0;
  float m[8], l[8];
#pragma unroll
  for (int r = 0; r < 8; ++r) { m[r] = -1e30f; l[r] = 0.f; }

  const unsigned short* kbase = kb + (size_t)bh * SEQ * DHEADD;
  const unsigned short* vbase = vb + (size_t)bh * DHEADD * SEQ;
  unsigned short* myP = lds[wave];

  for (int j0 = 0; j0 < SEQ; j0 += 32) {
    // --- scores: two 16x16 S tiles, K = 64 (2 WMMA each) ---
    const unsigned short* krow0 = kbase + (size_t)(j0 +      lr) * DHEADD;
    const unsigned short* krow1 = kbase + (size_t)(j0 + 16 + lr) * DHEADD;
    v8f s0 = {0.f,0.f,0.f,0.f,0.f,0.f,0.f,0.f};
    v8f s1 = s0;
    s0 = wmma_bf16(aq0, *(const v16bf*)(krow0 +      half * 16), s0);
    s0 = wmma_bf16(aq1, *(const v16bf*)(krow0 + 32 + half * 16), s0);
    s1 = wmma_bf16(aq0, *(const v16bf*)(krow1 +      half * 16), s1);
    s1 = wmma_bf16(aq1, *(const v16bf*)(krow1 + 32 + half * 16), s1);

    // --- online softmax (rows live on fixed VGPR slot across 16-lane half) ---
    float mx[8], corr[8], rs[8];
#pragma unroll
    for (int r = 0; r < 8; ++r) mx[r] = fmaxf(s0[r], s1[r]);
#pragma unroll
    for (int mask = 1; mask <= 8; mask <<= 1)
#pragma unroll
      for (int r = 0; r < 8; ++r) mx[r] = fmaxf(mx[r], __shfl_xor(mx[r], mask, 32));
#pragma unroll
    for (int r = 0; r < 8; ++r) {
      float mn = fmaxf(m[r], mx[r]);
      corr[r] = __expf(m[r] - mn);
      m[r] = mn;
      float p0 = __expf(s0[r] - mn);
      float p1 = __expf(s1[r] - mn);
      s0[r] = p0; s1[r] = p1;
      rs[r] = p0 + p1;
    }
#pragma unroll
    for (int mask = 1; mask <= 8; mask <<= 1)
#pragma unroll
      for (int r = 0; r < 8; ++r) rs[r] += __shfl_xor(rs[r], mask, 32);
#pragma unroll
    for (int r = 0; r < 8; ++r) {
      l[r] = l[r] * corr[r] + rs[r];
      acc0[r] *= corr[r]; acc1[r] *= corr[r];
      acc2[r] *= corr[r]; acc3[r] *= corr[r];
    }

    // --- P: C-layout -> LDS -> A-layout (16 x 32, K = j local) ---
#pragma unroll
    for (int r = 0; r < 8; ++r) {
      int row = r + half * 8;
      myP[row * 32 +      lr] = f32_to_bf16(s0[r]);
      myP[row * 32 + 16 + lr] = f32_to_bf16(s1[r]);
    }
    v16bf ap = join8(*(const v8bf*)(myP + lr * 32 + half * 16),
                     *(const v8bf*)(myP + lr * 32 + half * 16 + 8));

    // --- O += P * V, four 16-wide d chunks ---
    const unsigned short* vj = vbase + j0 + half * 16;
    acc0 = wmma_bf16(ap, *(const v16bf*)(vj + (size_t)( 0 + lr) * SEQ), acc0);
    acc1 = wmma_bf16(ap, *(const v16bf*)(vj + (size_t)(16 + lr) * SEQ), acc1);
    acc2 = wmma_bf16(ap, *(const v16bf*)(vj + (size_t)(32 + lr) * SEQ), acc2);
    acc3 = wmma_bf16(ap, *(const v16bf*)(vj + (size_t)(48 + lr) * SEQ), acc3);
  }

  // --- finalize: O /= l, store bf16 O^T[b][i][h*64 + d] ---
#pragma unroll
  for (int r = 0; r < 8; ++r) {
    int row = i0 + r + half * 8;
    float inv = 1.0f / l[r];
    size_t ob = ((size_t)b * SEQ + row) * INNERD + h * 64 + lr;
    outT[ob +  0] = f32_to_bf16(acc0[r] * inv);
    outT[ob + 16] = f32_to_bf16(acc1[r] * inv);
    outT[ob + 32] = f32_to_bf16(acc2[r] * inv);
    outT[ob + 48] = f32_to_bf16(acc3[r] * inv);
  }
}

// ---------------------------------------------------------------------------
// 6) Output projection: y[b][o][n] = sum_c w_out[o][c] * out[c][n] + b_out[o]
__global__ void __launch_bounds__(256) k_gemm_out(const unsigned short* __restrict__ wo,
                                                  const unsigned short* __restrict__ outT,
                                                  const float* __restrict__ bias,
                                                  float* __restrict__ y) {
  const int MT = DIMM / 64, NT = SEQ / 64;               // 16 x 32 macro-tiles
  int wid = __builtin_amdgcn_readfirstlane(
      blockIdx.x * ((int)blockDim.x >> 5) + ((int)threadIdx.x >> 5));
  int lane = threadIdx.x & 31, half = lane >> 4, lr = lane & 15;
  int b   = wid / (MT * NT);
  int rem = wid % (MT * NT);
  int m0  = (rem / NT) * 64, n0 = (rem % NT) * 64;

  v8f acc[4][4];
#pragma unroll
  for (int mi = 0; mi < 4; ++mi)
#pragma unroll
    for (int ni = 0; ni < 4; ++ni)
      acc[mi][ni] = (v8f){0.f,0.f,0.f,0.f,0.f,0.f,0.f,0.f};

  mma_block64(wo, INNERD, outT + (size_t)b * SEQ * INNERD, INNERD,
              INNERD, m0, n0, lr, half, acc);

#pragma unroll
  for (int mi = 0; mi < 4; ++mi) {
#pragma unroll
    for (int ni = 0; ni < 4; ++ni) {
      int n = n0 + ni * 16 + lr;
#pragma unroll
      for (int r = 0; r < 8; ++r) {
        int o = m0 + mi * 16 + r + half * 8;
        y[((size_t)b * DIMM + o) * SEQ + n] = acc[mi][ni][r] + bias[o];
      }
    }
  }
}

// ---------------------------------------------------------------------------
extern "C" void kernel_launch(void* const* d_in, const int* in_sizes, int n_in,
                              void* d_out, int out_size, void* d_ws, size_t ws_size,
                              hipStream_t stream) {
  (void)in_sizes; (void)n_in; (void)out_size; (void)ws_size;
  const float* x     = (const float*)d_in[0];   // [B][1024][2048]
  const float* w_qkv = (const float*)d_in[1];   // [3072][1024]
  const float* w_out = (const float*)d_in[2];   // [1024][1024]
  const float* b_out = (const float*)d_in[3];   // [1024]
  float* y = (float*)d_out;                      // [B][1024][2048]

  char* ws = (char*)d_ws;
  unsigned short* xT   = (unsigned short*)ws; ws += (size_t)BATCH * SEQ * DIMM * 2;
  unsigned short* wqb  = (unsigned short*)ws; ws += (size_t)QKVROWS * DIMM * 2;
  unsigned short* wob  = (unsigned short*)ws; ws += (size_t)DIMM * INNERD * 2;
  float*          qTf  = (float*)ws;          ws += (size_t)BATCH * NHEADS * SEQ * DHEADD * 4;
  float*          kTf  = (float*)ws;          ws += (size_t)BATCH * NHEADS * SEQ * DHEADD * 4;
  unsigned short* qbb  = (unsigned short*)ws; ws += (size_t)BATCH * NHEADS * SEQ * DHEADD * 2;
  unsigned short* kbb  = (unsigned short*)ws; ws += (size_t)BATCH * NHEADS * SEQ * DHEADD * 2;
  unsigned short* vbb  = (unsigned short*)ws; ws += (size_t)BATCH * NHEADS * DHEADD * SEQ * 2;
  unsigned short* outT = (unsigned short*)ws; ws += (size_t)BATCH * SEQ * INNERD * 2;

  // weight conversions
  k_cvt<<<(QKVROWS * DIMM) / 256, 256, 0, stream>>>(w_qkv, wqb, QKVROWS * DIMM);
  k_cvt<<<(DIMM * INNERD) / 256, 256, 0, stream>>>(w_out, wob, DIMM * INNERD);
  // x transpose + convert
  k_xT<<<(BATCH * SEQ * DIMM) / 256, 256, 0, stream>>>(x, xT);
  // QKV projection (64x64 macro-tile per wave)
  {
    int waves = BATCH * (QKVROWS / 64) * (SEQ / 64);     // 3072
    k_gemm_qkv<<<waves / 8, 256, 0, stream>>>(wqb, xT, qTf, kTf, vbb);
  }
  // RoPE + bf16
  k_rope<<<(BATCH * NHEADS * SEQ * DHEADD) / 256, 256, 0, stream>>>(qTf, kTf, qbb, kbb);
  // flash attention
  k_attn<<<(BATCH * NHEADS) * (SEQ / 64), 128, 0, stream>>>(qbb, kbb, vbb, outT);
  // output projection + bias (64x64 macro-tile per wave)
  {
    int waves = BATCH * (DIMM / 64) * (SEQ / 64);        // 1024
    k_gemm_out<<<waves / 8, 256, 0, stream>>>(wob, outT, b_out, y);
  }
}
